// ProposalGenerationNetwork_44839458570397
// MI455X (gfx1250) — matrix operand order repaired and checked by reading, not compiled, not run on hardware
//
#include <hip/hip_runtime.h>
#include <math.h>

typedef __attribute__((ext_vector_type(16))) _Float16 v16h;
typedef __attribute__((ext_vector_type(8)))  float    v8f;

#define HF 100
#define WFE 100
#define NSPAT (HF*WFE)          // 10000
#define CIN 1024
#define COUT 1024
#define NANCH 9
#define NBOX (NSPAT*NANCH)      // 90000
#define PRE_TOPK 6000
#define POST_TOPK 300
#define NSORT1 131072
#define NSORT2 8192
#define NMS_WORDS 188           // ceil(6000/32)
#define BBOX_CLIP_F 4.135166556742356f

// ---------------- pack / convert kernels ----------------

__global__ void k_convert_feat(const float* __restrict__ fm, _Float16* __restrict__ Ff) {
    int i = blockIdx.x * blockDim.x + threadIdx.x;   // over CIN*NSPAT, input order [c][s]
    if (i >= CIN * NSPAT) return;
    int c = i / NSPAT;
    int s = i - c * NSPAT;
    Ff[(size_t)s * CIN + c] = (_Float16)fm[i];       // channels-last
}

// conv weights: input [co][ci][kyx] -> output [kyx][co][ci]  (ci innermost for LDS B-transpose)
__global__ void k_pack_conv_w(const float* __restrict__ w, _Float16* __restrict__ Wt) {
    int i = blockIdx.x * blockDim.x + threadIdx.x;   // output-linear for coalesced writes
    if (i >= 9 * COUT * CIN) return;
    int ci  = i % CIN;
    int co  = (i / CIN) % COUT;
    int kyx = i / (CIN * COUT);
    Wt[i] = (_Float16)w[((size_t)co * CIN + ci) * 9 + kyx];
}

// head weights: [n][ci], n: 0..8 obj, 9..44 reg, 45..63 zero
__global__ void k_pack_head_w(const float* __restrict__ ow, const float* __restrict__ rw,
                              _Float16* __restrict__ Wh) {
    int i = blockIdx.x * blockDim.x + threadIdx.x;   // over 64*CIN
    if (i >= 64 * CIN) return;
    int n = i / CIN;
    float v = 0.f;
    if (n < 9)       v = ow[i];
    else if (n < 45) v = rw[i - 9 * CIN];
    Wh[i] = (_Float16)v;
}

// ---------------- WMMA 3x3 conv (implicit GEMM) ----------------
// M=10000 spatial, N=1024 cout, K=9*1024.
// Block tile 64M x 256N, 8 waves (2M x 4N), wave tile 32x64 (acc[2][4]).

__global__ __launch_bounds__(256) void k_conv3x3_wmma(
    const _Float16* __restrict__ Ff,   // [NSPAT][CIN] channels-last
    const _Float16* __restrict__ Wt,   // [9][COUT][CIN]
    const float*    __restrict__ bias, // [COUT]
    _Float16*       __restrict__ outF) // [NSPAT][COUT]
{
    __shared__ _Float16 As[64][40];    // [m][k]
    __shared__ _Float16 Bs[256][40];   // [n][k]  (transposed: contiguous-k per n)

    const int tid  = threadIdx.x;
    const int lane = tid & 31;
    const int wid  = tid >> 5;
    const int waveM = wid & 1;          // 2 x 32 rows
    const int waveN = wid >> 1;         // 4 x 64 cols
    const int m0 = blockIdx.x * 64;
    const int n0 = blockIdx.y * 256;

    v8f zero = {0.f,0.f,0.f,0.f,0.f,0.f,0.f,0.f};
    v8f acc[2][4];
#pragma unroll
    for (int i = 0; i < 2; ++i)
#pragma unroll
        for (int j = 0; j < 4; ++j) acc[i][j] = zero;

    const int ar = (tid * 8) >> 5;      // A stage: row 0..63
    const int ac = (tid * 8) & 31;      // col 0,8,16,24
    const int am = m0 + ar;
    const int ay = am / WFE;
    const int ax = am - ay * WFE;

    const int frow = lane & 15;
    const int kbA  = (lane >> 4) << 4;  // A frag: K 0-15 vs 16-31 (contiguous)
    const int kbB  = (lane >> 4) << 4;  // B frag: K 0-15 vs 16-31 (contiguous)

    for (int kyx = 0; kyx < 9; ++kyx) {
        const int ky = kyx / 3 - 1;
        const int kx = kyx - (kyx / 3) * 3 - 1;
        const int yy = ay + ky;
        const int xx = ax + kx;
        const bool inb = (am < NSPAT) && (yy >= 0) && (yy < HF) && (xx >= 0) && (xx < WFE);
        const _Float16* fsrc = inb ? (Ff + ((size_t)(yy * WFE + xx)) * CIN) : Ff;
        const _Float16* wsrc = Wt + ((size_t)kyx * COUT + n0 + tid) * CIN;  // B row = tid

        for (int c0 = 0; c0 < CIN; c0 += 32) {
            // stage A tile (8 halfs per thread)
            if (inb) {
                *(uint4*)&As[ar][ac] = *(const uint4*)(fsrc + c0 + ac);
            } else {
                uint4 z = {0,0,0,0};
                *(uint4*)&As[ar][ac] = z;
            }
            // stage B tile: each thread stages its own n-row (32 halfs)
            {
                const uint4* p = (const uint4*)(wsrc + c0);
                *(uint4*)&Bs[tid][0]  = p[0];
                *(uint4*)&Bs[tid][8]  = p[1];
                *(uint4*)&Bs[tid][16] = p[2];
                *(uint4*)&Bs[tid][24] = p[3];
            }
            __syncthreads();

            v16h afrag[2], bfrag[4];
#pragma unroll
            for (int i = 0; i < 2; ++i) {
                const int mr = waveM * 32 + i * 16 + frow;
#pragma unroll
                for (int e = 0; e < 16; ++e) {
                    const int k = kbA + e;
                    afrag[i][e] = As[mr][k];
                }
            }
#pragma unroll
            for (int j = 0; j < 4; ++j) {
                const int nc = waveN * 64 + j * 16 + frow;
#pragma unroll
                for (int e = 0; e < 16; ++e) {
                    bfrag[j][e] = Bs[nc][kbB + e];      // contiguous -> ds_load_b128 x2
                }
            }
#pragma unroll
            for (int i = 0; i < 2; ++i)
#pragma unroll
                for (int j = 0; j < 4; ++j)
                    acc[i][j] = __builtin_amdgcn_wmma_f32_16x16x32_f16(
                        false, afrag[i], false, bfrag[j], (short)0, acc[i][j], false, false);
            __syncthreads();
        }
    }

    const int cm = (lane >> 4) << 3;
    const int cn = lane & 15;
#pragma unroll
    for (int i = 0; i < 2; ++i) {
#pragma unroll
        for (int j = 0; j < 4; ++j) {
            const int n = n0 + waveN * 64 + j * 16 + cn;
            const float bv = bias[n];
#pragma unroll
            for (int r = 0; r < 8; ++r) {
                const int m = m0 + waveM * 32 + i * 16 + cm + r;
                if (m < NSPAT) {
                    float v = acc[i][j][r] + bv;
                    outF[(size_t)m * COUT + n] = (_Float16)(v > 0.f ? v : 0.f);
                }
            }
        }
    }
}

// ---------------- WMMA head GEMM: M=10000, K=1024, N=64 ----------------
// Block: 128M x 64N, 8 waves (4M x 2N), wave = 32x32.

__global__ __launch_bounds__(256) void k_head_wmma(
    const _Float16* __restrict__ feats, // [NSPAT][CIN]
    const _Float16* __restrict__ Wh,    // [64][CIN]
    float* __restrict__ outH)           // [NSPAT][64]
{
    __shared__ _Float16 As[128][40];   // [m][k]
    __shared__ _Float16 Bs[64][40];    // [n][k] transposed

    const int tid  = threadIdx.x;
    const int lane = tid & 31;
    const int wid  = tid >> 5;
    const int waveM = wid & 3;          // 4 x 32 rows
    const int waveN = wid >> 2;         // 2 x 32 cols
    const int m0 = blockIdx.x * 128;

    v8f zero = {0.f,0.f,0.f,0.f,0.f,0.f,0.f,0.f};
    v8f acc[2][2];
    acc[0][0]=zero; acc[0][1]=zero; acc[1][0]=zero; acc[1][1]=zero;

    const int ar = tid >> 1;            // 0..127
    const int ac = (tid & 1) << 4;      // 0 or 16
    const int am = m0 + ar;
    const bool ain = am < NSPAT;

    const int brn = tid >> 2;           // B stage: n row 0..63
    const int brk = (tid & 3) << 3;     // k col 0,8,16,24

    const int frow = lane & 15;
    const int kbA  = (lane >> 4) << 4;
    const int kbB  = (lane >> 4) << 4;

    for (int c0 = 0; c0 < CIN; c0 += 32) {
        if (ain) {
            const uint4* s = (const uint4*)(feats + (size_t)am * CIN + c0 + ac);
            *(uint4*)&As[ar][ac]     = s[0];
            *(uint4*)&As[ar][ac + 8] = s[1];
        } else {
            uint4 z = {0,0,0,0};
            *(uint4*)&As[ar][ac] = z;
            *(uint4*)&As[ar][ac + 8] = z;
        }
        *(uint4*)&Bs[brn][brk] = *(const uint4*)(Wh + (size_t)brn * CIN + c0 + brk);
        __syncthreads();

        v16h afrag[2], bfrag[2];
#pragma unroll
        for (int i = 0; i < 2; ++i) {
            const int mr = waveM * 32 + i * 16 + frow;
#pragma unroll
            for (int e = 0; e < 16; ++e) {
                const int k = kbA + e;
                afrag[i][e] = As[mr][k];
            }
        }
#pragma unroll
        for (int j = 0; j < 2; ++j) {
            const int nc = waveN * 32 + j * 16 + frow;
#pragma unroll
            for (int e = 0; e < 16; ++e) {
                bfrag[j][e] = Bs[nc][kbB + e];
            }
        }
#pragma unroll
        for (int i = 0; i < 2; ++i)
#pragma unroll
            for (int j = 0; j < 2; ++j)
                acc[i][j] = __builtin_amdgcn_wmma_f32_16x16x32_f16(
                    false, afrag[i], false, bfrag[j], (short)0, acc[i][j], false, false);
        __syncthreads();
    }

    const int cm = (lane >> 4) << 3;
    const int cn = lane & 15;
#pragma unroll
    for (int i = 0; i < 2; ++i) {
#pragma unroll
        for (int j = 0; j < 2; ++j) {
            const int n = waveN * 32 + j * 16 + cn;
#pragma unroll
            for (int r = 0; r < 8; ++r) {
                const int m = m0 + waveM * 32 + i * 16 + cm + r;
                if (m < NSPAT) outH[(size_t)m * 64 + n] = acc[i][j][r];
            }
        }
    }
}

// ---------------- decode + sigmoid + sort-key build ----------------

__global__ void k_decode(const float* __restrict__ outH, const float* __restrict__ ob,
                         const float* __restrict__ rb, float* __restrict__ boxes,
                         unsigned long long* __restrict__ keys)
{
    int i = blockIdx.x * blockDim.x + threadIdx.x;
    if (i >= NSORT1) return;
    if (i >= NBOX) { keys[i] = 0ull; return; }
    int m = i / NANCH;
    int a = i - m * NANCH;

    float obj = outH[(size_t)m * 64 + a] + ob[a];
    float s = 1.0f / (1.0f + expf(-obj));

    float dx = outH[(size_t)m * 64 + 9 + a * 4 + 0] + rb[a * 4 + 0];
    float dy = outH[(size_t)m * 64 + 9 + a * 4 + 1] + rb[a * 4 + 1];
    float dw = fminf(outH[(size_t)m * 64 + 9 + a * 4 + 2] + rb[a * 4 + 2], BBOX_CLIP_F);
    float dh = fminf(outH[(size_t)m * 64 + 9 + a * 4 + 3] + rb[a * 4 + 3], BBOX_CLIP_F);

    int ri = a / 3, si = a - ri * 3;
    float ratio = (ri == 0) ? 0.5f : ((ri == 1) ? 1.0f : 2.0f);
    float scale = (si == 0) ? 128.f : ((si == 1) ? 256.f : 512.f);
    float hr = sqrtf(ratio);
    float wa = scale / hr, ha = scale * hr;
    float cxa = (float)(m % WFE) * 16.f;
    float cya = (float)(m / WFE) * 16.f;

    float cx = dx * wa + cxa, cy = dy * ha + cya;
    float w = expf(dw) * wa, h = expf(dh) * ha;

    boxes[(size_t)i * 4 + 0] = cx - 0.5f * w;
    boxes[(size_t)i * 4 + 1] = cy - 0.5f * h;
    boxes[(size_t)i * 4 + 2] = cx + 0.5f * w;
    boxes[(size_t)i * 4 + 3] = cy + 0.5f * h;

    unsigned int b = __float_as_uint(s);
    b = (b & 0x80000000u) ? ~b : (b | 0x80000000u);   // order-preserving float map
    keys[i] = ((unsigned long long)b << 32) | (unsigned long long)(0xFFFFFFFFu - (unsigned)i);
}

// ---------------- bitonic sort (descending) ----------------

__global__ void k_bitonic(unsigned long long* __restrict__ keys, int j, int k, int n)
{
    int i = blockIdx.x * blockDim.x + threadIdx.x;
    int ixj = i ^ j;
    if (ixj > i && ixj < n && i < n) {
        unsigned long long a = keys[i], b = keys[ixj];
        bool doSwap = ((i & k) == 0) ? (a < b) : (a > b);   // descending overall
        if (doSwap) { keys[i] = b; keys[ixj] = a; }
    }
}

// ---------------- gather top-6000, clip, valid ----------------

__global__ void k_gather(const unsigned long long* __restrict__ keys,
                         const float* __restrict__ boxes,
                         float* __restrict__ tb, float* __restrict__ ts,
                         unsigned int* __restrict__ valid)
{
    int t = blockIdx.x * blockDim.x + threadIdx.x;
    if (t >= PRE_TOPK) return;
    unsigned long long key = keys[t];
    unsigned int idx = 0xFFFFFFFFu - (unsigned int)(key & 0xFFFFFFFFull);
    unsigned int sb = (unsigned int)(key >> 32);
    unsigned int fb = (sb & 0x80000000u) ? (sb ^ 0x80000000u) : ~sb;
    float s = __uint_as_float(fb);
    float x0 = 0.f, y0 = 0.f, x1 = 0.f, y1 = 0.f;
    if (idx < NBOX) {
        x0 = fminf(fmaxf(boxes[(size_t)idx * 4 + 0], 0.f), 1600.f);
        y0 = fminf(fmaxf(boxes[(size_t)idx * 4 + 1], 0.f), 1600.f);
        x1 = fminf(fmaxf(boxes[(size_t)idx * 4 + 2], 0.f), 1600.f);
        y1 = fminf(fmaxf(boxes[(size_t)idx * 4 + 3], 0.f), 1600.f);
    }
    tb[t * 4 + 0] = x0; tb[t * 4 + 1] = y0; tb[t * 4 + 2] = x1; tb[t * 4 + 3] = y1;
    ts[t] = s;
    valid[t] = ((x1 - x0) >= 16.f && (y1 - y0) >= 16.f) ? 1u : 0u;
}

// ---------------- NMS: IoU bitmask (ballot) + sequential sweep ----------------

__global__ void k_nms_mask(const float* __restrict__ tb, unsigned int* __restrict__ mask)
{
    int i = blockIdx.x;           // 0..5999
    int w = blockIdx.y;           // 0..187
    int lane = threadIdx.x;       // 0..31
    int j = w * 32 + lane;
    bool p = false;
    float ax0 = tb[i * 4 + 0], ay0 = tb[i * 4 + 1], ax1 = tb[i * 4 + 2], ay1 = tb[i * 4 + 3];
    if (j < PRE_TOPK && j > i) {
        float bx0 = tb[j * 4 + 0], by0 = tb[j * 4 + 1], bx1 = tb[j * 4 + 2], by1 = tb[j * 4 + 3];
        float areaA = (ax1 - ax0) * (ay1 - ay0);
        float areaB = (bx1 - bx0) * (by1 - by0);
        float iw = fmaxf(fminf(ax1, bx1) - fmaxf(ax0, bx0), 0.f);
        float ih = fmaxf(fminf(ay1, by1) - fmaxf(ay0, by0), 0.f);
        float inter = iw * ih;
        float iou = inter / (areaA + areaB - inter);
        p = iou > 0.7f;
    }
    unsigned long long bal = __ballot(p);
    if (lane == 0) mask[(size_t)i * NMS_WORDS + w] = (unsigned int)bal;
}

__global__ __launch_bounds__(256) void k_nms_seq(const unsigned int* __restrict__ mask,
                                                 const unsigned int* __restrict__ valid,
                                                 unsigned int* __restrict__ keepW)
{
    __shared__ unsigned int kw[NMS_WORDS];
    int t = threadIdx.x;
    for (int w = t; w < NMS_WORDS; w += 256) {
        unsigned int v = 0;
        for (int b = 0; b < 32; ++b) {
            int j = w * 32 + b;
            if (j < PRE_TOPK && valid[j]) v |= (1u << b);
        }
        kw[w] = v;
    }
    __syncthreads();
    for (int i = 0; i < PRE_TOPK; ++i) {
        bool alive = (kw[i >> 5] >> (i & 31)) & 1u;
        __syncthreads();
        if (alive) {
            for (int w = t; w < NMS_WORDS; w += 256)
                kw[w] &= ~mask[(size_t)i * NMS_WORDS + w];
        }
        __syncthreads();
    }
    for (int w = t; w < NMS_WORDS; w += 256) keepW[w] = kw[w];
}

// ---------------- final top-300 ----------------

__global__ void k_mask_scores(const float* __restrict__ ts, const unsigned int* __restrict__ keepW,
                              unsigned long long* __restrict__ keys2)
{
    int i = blockIdx.x * blockDim.x + threadIdx.x;
    if (i >= NSORT2) return;
    if (i >= PRE_TOPK) { keys2[i] = 0ull; return; }
    bool alive = (keepW[i >> 5] >> (i & 31)) & 1u;
    float s = alive ? ts[i] : -1.0f;
    unsigned int b = __float_as_uint(s);
    b = (b & 0x80000000u) ? ~b : (b | 0x80000000u);
    keys2[i] = ((unsigned long long)b << 32) | (unsigned long long)(0xFFFFFFFFu - (unsigned)i);
}

__global__ void k_writeout(const unsigned long long* __restrict__ keys2,
                           const float* __restrict__ tb, float* __restrict__ out)
{
    int r = blockIdx.x * blockDim.x + threadIdx.x;
    if (r >= POST_TOPK) return;
    unsigned long long key = keys2[r];
    unsigned int t = 0xFFFFFFFFu - (unsigned int)(key & 0xFFFFFFFFull);
    unsigned int sb = (unsigned int)(key >> 32);
    unsigned int fb = (sb & 0x80000000u) ? (sb ^ 0x80000000u) : ~sb;
    float s = __uint_as_float(fb);
    bool good = (s > 0.f) && (t < PRE_TOPK);
    out[r * 4 + 0] = good ? tb[t * 4 + 0] : 0.f;
    out[r * 4 + 1] = good ? tb[t * 4 + 1] : 0.f;
    out[r * 4 + 2] = good ? tb[t * 4 + 2] : 0.f;
    out[r * 4 + 3] = good ? tb[t * 4 + 3] : 0.f;
    out[POST_TOPK * 4 + r] = good ? s : 0.f;
}

// ---------------- launch ----------------

extern "C" void kernel_launch(void* const* d_in, const int* in_sizes, int n_in,
                              void* d_out, int out_size, void* d_ws, size_t ws_size,
                              hipStream_t stream)
{
    const float* fm     = (const float*)d_in[1];
    const float* conv_w = (const float*)d_in[2];
    const float* conv_b = (const float*)d_in[3];
    const float* obj_w  = (const float*)d_in[4];
    const float* obj_b  = (const float*)d_in[5];
    const float* reg_w  = (const float*)d_in[6];
    const float* reg_b  = (const float*)d_in[7];
    float* out = (float*)d_out;

    char* base = (char*)d_ws;
    size_t off = 0;
    auto take = [&](size_t bytes) -> void* {
        void* p = base + off;
        off = (off + bytes + 255) & ~(size_t)255;
        return p;
    };
    _Float16* Ff     = (_Float16*)take((size_t)NSPAT * CIN * 2);
    _Float16* Wt     = (_Float16*)take((size_t)9 * COUT * CIN * 2);
    _Float16* Wh     = (_Float16*)take((size_t)64 * CIN * 2);
    _Float16* feats2 = (_Float16*)take((size_t)NSPAT * COUT * 2);
    float* outH      = (float*)take((size_t)NSPAT * 64 * 4);
    float* boxes     = (float*)take((size_t)NBOX * 4 * 4);
    unsigned long long* keys = (unsigned long long*)take((size_t)NSORT1 * 8);
    float* tb        = (float*)take((size_t)PRE_TOPK * 4 * 4);
    float* ts        = (float*)take((size_t)PRE_TOPK * 4);
    unsigned int* valid = (unsigned int*)take((size_t)PRE_TOPK * 4);
    unsigned int* mask  = (unsigned int*)take((size_t)PRE_TOPK * NMS_WORDS * 4);
    unsigned int* keepW = (unsigned int*)take((size_t)NMS_WORDS * 4);
    unsigned long long* keys2 = (unsigned long long*)take((size_t)NSORT2 * 8);

    { int n = CIN * NSPAT;    k_convert_feat<<<(n + 255) / 256, 256, 0, stream>>>(fm, Ff); }
    { int n = 9 * COUT * CIN; k_pack_conv_w<<<(n + 255) / 256, 256, 0, stream>>>(conv_w, Wt); }
    { int n = 64 * CIN;       k_pack_head_w<<<(n + 255) / 256, 256, 0, stream>>>(obj_w, reg_w, Wh); }

    {
        dim3 grid((NSPAT + 63) / 64, COUT / 256);
        k_conv3x3_wmma<<<grid, 256, 0, stream>>>(Ff, Wt, conv_b, feats2);
    }
    k_head_wmma<<<(NSPAT + 127) / 128, 256, 0, stream>>>(feats2, Wh, outH);

    k_decode<<<NSORT1 / 256, 256, 0, stream>>>(outH, obj_b, reg_b, boxes, keys);

    for (int k = 2; k <= NSORT1; k <<= 1)
        for (int j = k >> 1; j > 0; j >>= 1)
            k_bitonic<<<NSORT1 / 256, 256, 0, stream>>>(keys, j, k, NSORT1);

    k_gather<<<(PRE_TOPK + 255) / 256, 256, 0, stream>>>(keys, boxes, tb, ts, valid);

    {
        dim3 grid(PRE_TOPK, NMS_WORDS);
        k_nms_mask<<<grid, 32, 0, stream>>>(tb, mask);
    }
    k_nms_seq<<<1, 256, 0, stream>>>(mask, valid, keepW);

    k_mask_scores<<<NSORT2 / 256, 256, 0, stream>>>(ts, keepW, keys2);

    for (int k = 2; k <= NSORT2; k <<= 1)
        for (int j = k >> 1; j > 0; j >>= 1)
            k_bitonic<<<NSORT2 / 256, 256, 0, stream>>>(keys2, j, k, NSORT2);

    k_writeout<<<(POST_TOPK + 255) / 256, 256, 0, stream>>>(keys2, tb, out);

    (void)in_sizes; (void)n_in; (void)out_size; (void)ws_size;
}
